// MaskedMultiHeadAttention_61314953118087
// MI455X (gfx1250) — matrix operand order repaired and checked by reading, not compiled, not run on hardware
//
#include <hip/hip_runtime.h>

// ---------------------------------------------------------------------------
// MaskedMultiHeadAttention for MI455X (gfx1250, wave32, WMMA bf16)
// B=4, S=2048, D=1024, H=16, HD=64
// ---------------------------------------------------------------------------

typedef __attribute__((ext_vector_type(16))) __bf16 v16bf;
typedef __attribute__((ext_vector_type(8)))  float  v8f;

#define DEV static __device__ __forceinline__

constexpr int kB  = 4;
constexpr int kS  = 2048;
constexpr int kD  = 1024;
constexpr int kH  = 16;
constexpr int kHD = 64;
constexpr int kM  = kB * kS;     // 8192 token rows

DEV unsigned short f2bf(float f) {
    unsigned int u = __float_as_uint(f);
    u += 0x7FFFu + ((u >> 16) & 1u);           // round-to-nearest-even
    return (unsigned short)(u >> 16);
}

// B-fragment: lane = one column, 16 contiguous K elements (32 bytes)
DEV v16bf load_b16(const unsigned short* p) {
    v16bf v; __builtin_memcpy(&v, p, 32); return v;
}
// A-fragment: lane = one row; elems 0..7 at K=16*half, elems 8..15 at K=16*half+8
DEV v16bf make_a16(const unsigned short* p0, const unsigned short* p1) {
    v16bf v;
    __builtin_memcpy(&v, p0, 16);
    __builtin_memcpy(reinterpret_cast<char*>(&v) + 16, p1, 16);
    return v;
}
DEV v8f v8zero() {
    v8f v;
#pragma unroll
    for (int i = 0; i < 8; ++i) v[i] = 0.0f;
    return v;
}
DEV v8f wmma_bf16(v16bf a, v16bf b, v8f c) {
    return __builtin_amdgcn_wmma_f32_16x16x32_bf16(
        /*neg_a=*/false, a, /*neg_b=*/false, b,
        /*c_mod=*/(short)0, c, /*reuse_a=*/false, /*reuse_b=*/false);
}
DEV float hmax16(float v) {
#pragma unroll
    for (int m = 1; m < 16; m <<= 1) v = fmaxf(v, __shfl_xor(v, m, 16));
    return v;
}
DEV float hsum16(float v) {
#pragma unroll
    for (int m = 1; m < 16; m <<= 1) v += __shfl_xor(v, m, 16);
    return v;
}

// gfx1250 async copy: global -> LDS, tracked by ASYNCcnt (no VGPR round-trip)
DEV void async_load_b128(unsigned lds_off, const void* gaddr) {
    asm volatile("global_load_async_to_lds_b128 %0, %1, off"
                 :: "v"(lds_off), "v"(gaddr) : "memory");
}
DEV void wait_asynccnt0() {
    asm volatile("s_wait_asynccnt 0" ::: "memory");
}

// ---------------------------------------------------------------------------
// GEMM: out[128x64 block tile] = A[M x 1024] @ W[1024 x 1024] + bias, *oscale
// MODE 0: A=f32 x, out=bf16 Q/K as [B,H,S,HD]   (oscale folds 1/sqrt(HD) for Q)
// MODE 1: A=f32 x, out=bf16 V transposed as [B,H,HD,S]
// MODE 2: A=bf16 ctx (async-staged), out=f32 [M,D] (+bias)
// Block: 128 threads = 4 waves; each wave computes 32 rows x 64 cols (8 WMMA/k).
// ---------------------------------------------------------------------------
template<int MODE>
__global__ __launch_bounds__(128)
void gemm_kernel(const float* __restrict__ Af,
                 const unsigned short* __restrict__ Ab,
                 const float* __restrict__ W,
                 const float* __restrict__ bias,
                 float oscale,
                 void* __restrict__ outp) {
    __shared__ unsigned short a_lds[128][40];  // [m][k], padded rows (80 B)
    __shared__ unsigned short w_lds[64][40];   // [n][k] (transposed W tile)

    const int tid  = threadIdx.x;
    const int wid  = tid >> 5;
    const int lane = tid & 31;
    const int nl   = lane & 15;
    const int hh   = lane >> 4;
    const int mbase = blockIdx.x * 128;
    const int nbase = blockIdx.y * 64;

    v8f acc[2][4];
#pragma unroll
    for (int g = 0; g < 2; ++g)
#pragma unroll
        for (int t = 0; t < 4; ++t) acc[g][t] = v8zero();

    for (int kb = 0; kb < kD; kb += 32) {
        // ---- stage A tile [128 x 32] as bf16 ----
        if (MODE == 2) {
            // ctx is already bf16: pure byte copy -> async DMA to LDS.
            // 128 rows x 64 B = 512 16-byte chunks; 4 per thread.
#pragma unroll
            for (int idx = tid; idx < 512; idx += 128) {
                const int r = idx >> 2, c = (idx & 3) * 8;
                async_load_b128((unsigned)(uintptr_t)&a_lds[r][c],
                                Ab + (size_t)(mbase + r) * kD + kb + c);
            }
        } else {
#pragma unroll 4
            for (int idx = tid; idx < 128 * 32; idx += 128) {
                const int r = idx >> 5, c = idx & 31;
                a_lds[r][c] = f2bf(Af[(size_t)(mbase + r) * kD + kb + c]);
            }
        }
        // ---- stage W tile transposed [n][k] (coalesced along n) ----
#pragma unroll 4
        for (int idx = tid; idx < 32 * 64; idx += 128) {
            const int kk = idx >> 6, n = idx & 63;
            w_lds[n][kk] = f2bf(W[(size_t)(kb + kk) * kD + nbase + n]);
        }
        if (kb + 32 < kD)    // prefetch next W K-slab (global_prefetch_b8)
            __builtin_prefetch(&W[(size_t)(kb + 32 + (tid >> 2)) * kD + nbase], 0, 0);
        if (MODE == 2) wait_asynccnt0();
        __syncthreads();

        v16bf a[2];
#pragma unroll
        for (int g = 0; g < 2; ++g) {
            const unsigned short* arow = &a_lds[wid * 32 + g * 16 + nl][0];
            a[g] = make_a16(arow + 16 * hh, arow + 16 * hh + 8);
        }
#pragma unroll
        for (int t = 0; t < 4; ++t) {
            const v16bf b = load_b16(&w_lds[t * 16 + nl][16 * hh]);
            acc[0][t] = wmma_bf16(a[0], b, acc[0][t]);
            acc[1][t] = wmma_bf16(a[1], b, acc[1][t]);
        }
        __syncthreads();
    }

    // ---- epilogue: C/D layout lane (n = nl, m = r + 8*half) ----
#pragma unroll
    for (int g = 0; g < 2; ++g) {
#pragma unroll
        for (int t = 0; t < 4; ++t) {
#pragma unroll
            for (int r = 0; r < 8; ++r) {
                const int m = mbase + wid * 32 + g * 16 + r + 8 * hh;
                const int n = nbase + t * 16 + nl;
                const float v = (acc[g][t][r] + bias[n]) * oscale;
                if (MODE == 2) {
                    ((float*)outp)[(size_t)m * kD + n] = v;
                } else {
                    const int b = m >> 11, s = m & (kS - 1);
                    const int h = n >> 6, hd = n & (kHD - 1);
                    unsigned short* O = (unsigned short*)outp;
                    if (MODE == 0)      // [B,H,S,HD]
                        O[(((size_t)(b * kH + h) * kS + s) << 6) + hd] = f2bf(v);
                    else                // [B,H,HD,S]
                        O[(((size_t)(b * kH + h) * kHD + hd) << 11) + s] = f2bf(v);
                }
            }
        }
    }
}

// ---------------------------------------------------------------------------
// Flash attention: one wave per 16-row q tile; 32 keys / step; online softmax.
// Q (pre-scaled by 1/sqrt(HD)), K: bf16 [B,H,S,HD]; Vt: bf16 [B,H,HD,S]
// Ctx out: bf16 [B,S,D]
// ---------------------------------------------------------------------------
__global__ __launch_bounds__(128)
void attn_kernel(const unsigned short* __restrict__ Q,
                 const unsigned short* __restrict__ K,
                 const unsigned short* __restrict__ Vt,
                 unsigned short* __restrict__ Ctx) {
    __shared__ unsigned short p_lds[4][16][40];   // per-wave P bounce tile

    const int wid  = threadIdx.x >> 5;
    const int lane = threadIdx.x & 31;
    const int nl   = lane & 15;
    const int hh   = lane >> 4;

    const int tile  = blockIdx.x * 4 + wid;       // 0 .. B*H*(S/16)-1
    const int qt    = tile & (kS / 16 - 1);
    const int bh    = tile >> 7;
    const int qbase = qt * 16;
    const int b     = bh >> 4;
    const int h     = bh & (kH - 1);

    const unsigned short* Qb = Q  + (size_t)bh * kS * kHD;
    const unsigned short* Kb = K  + (size_t)bh * kS * kHD;
    const unsigned short* Vb = Vt + (size_t)bh * kHD * kS;

    // Q A-fragments for the whole tile (row m = qbase + nl), HD=64 -> 2 chunks
    v16bf aq[2];
    {
        const unsigned short* qrow = Qb + (size_t)(qbase + nl) * kHD;
#pragma unroll
        for (int c = 0; c < 2; ++c)
            aq[c] = make_a16(qrow + c * 32 + 16 * hh, qrow + c * 32 + 16 * hh + 8);
    }

    const float NEG = -1.0e30f;
    float m_i[8], l_i[8];
    v8f o[4];
#pragma unroll
    for (int r = 0; r < 8; ++r) { m_i[r] = NEG; l_i[r] = 0.0f; }
#pragma unroll
    for (int t = 0; t < 4; ++t) o[t] = v8zero();

    const int kv_end = qbase + 16;                // causal: keys < qbase+16

    for (int kv = 0; kv < kv_end; kv += 32) {
        // ---- scores: two 16x16 tiles over 32 keys, K-dim = HD=64 ----
        v8f s0 = v8zero(), s1 = v8zero();
#pragma unroll
        for (int c = 0; c < 2; ++c) {
            const v16bf bk0 = load_b16(Kb + (size_t)(kv + nl)      * kHD + c * 32 + 16 * hh);
            const v16bf bk1 = load_b16(Kb + (size_t)(kv + 16 + nl) * kHD + c * 32 + 16 * hh);
            s0 = wmma_bf16(aq[c], bk0, s0);
            s1 = wmma_bf16(aq[c], bk1, s1);
        }

        const bool boundary = (kv + 31 > qbase);
#pragma unroll
        for (int r = 0; r < 8; ++r) {
            float v0 = s0[r];                     // scale pre-folded into Q
            float v1 = s1[r];
            if (boundary) {
                const int m = qbase + r + 8 * hh;
                if (kv + nl      > m) v0 = NEG;
                if (kv + 16 + nl > m) v1 = NEG;
            }
            // ---- online softmax (rows live in 16-lane halves) ----
            const float rmax = hmax16(fmaxf(v0, v1));
            const float mnew = fmaxf(m_i[r], rmax);
            const float alpha = __expf(m_i[r] - mnew);
            const float p0 = __expf(v0 - mnew);
            const float p1 = __expf(v1 - mnew);
            l_i[r] = l_i[r] * alpha + hsum16(p0 + p1);
            m_i[r] = mnew;
#pragma unroll
            for (int t = 0; t < 4; ++t) o[t][r] *= alpha;
            // P -> LDS (C-layout scatter, row = r+8*half, col = key-local)
            p_lds[wid][r + 8 * hh][nl]      = f2bf(p0);
            p_lds[wid][r + 8 * hh][nl + 16] = f2bf(p1);
        }

        // ---- P(16x32) @ V(32x64): read P back in A-layout ----
        const unsigned short* prow = &p_lds[wid][nl][0];
        const v16bf pa = make_a16(prow + 16 * hh, prow + 16 * hh + 8);
#pragma unroll
        for (int t = 0; t < 4; ++t) {
            const v16bf bv = load_b16(Vb + (size_t)(t * 16 + nl) * kS + kv + 16 * hh);
            o[t] = wmma_bf16(pa, bv, o[t]);
        }
    }

    // ---- normalize and store ctx [B,S,D] bf16 ----
#pragma unroll
    for (int r = 0; r < 8; ++r) {
        const float inv = 1.0f / l_i[r];
        const int s = qbase + r + 8 * hh;
#pragma unroll
        for (int t = 0; t < 4; ++t) {
            const int n = h * kHD + t * 16 + nl;
            Ctx[((size_t)(b * kS + s) << 10) + n] = f2bf(o[t][r] * inv);
        }
    }
}

// ---------------------------------------------------------------------------
// Host launcher
// ---------------------------------------------------------------------------
extern "C" void kernel_launch(void* const* d_in, const int* in_sizes, int n_in,
                              void* d_out, int out_size, void* d_ws, size_t ws_size,
                              hipStream_t stream) {
    (void)in_sizes; (void)n_in; (void)out_size; (void)ws_size;
    const float* x  = (const float*)d_in[0];
    const float* wq = (const float*)d_in[1];
    const float* bq = (const float*)d_in[2];
    const float* wk = (const float*)d_in[3];
    const float* bk = (const float*)d_in[4];
    const float* wv = (const float*)d_in[5];
    const float* bv = (const float*)d_in[6];
    const float* wo = (const float*)d_in[7];
    const float* bo = (const float*)d_in[8];

    const size_t MN = (size_t)kM * kD;            // 8M elements
    unsigned short* qws = (unsigned short*)d_ws;
    unsigned short* kws = qws + MN;
    unsigned short* vws = kws + MN;
    unsigned short* cws = vws + MN;               // total 64 MB bf16 scratch

    dim3 gg(kM / 128, kD / 64);                   // 64 x 16 block tiles
    gemm_kernel<0><<<gg, 128, 0, stream>>>(x, nullptr, wq, bq, 0.125f, qws); // Q pre-scaled
    gemm_kernel<0><<<gg, 128, 0, stream>>>(x, nullptr, wk, bk, 1.0f,   kws);
    gemm_kernel<1><<<gg, 128, 0, stream>>>(x, nullptr, wv, bv, 1.0f,   vws);

    const int n_qtiles = kB * kH * (kS / 16);     // 8192 waves, 4 per block
    attn_kernel<<<n_qtiles / 4, 128, 0, stream>>>(qws, kws, vws, cws);

    gemm_kernel<2><<<gg, 128, 0, stream>>>(nullptr, cws, wo, bo, 1.0f, (float*)d_out);
}
